// NF4_34935263985943
// MI455X (gfx1250) — matrix operand order repaired and checked by reading, not compiled, not run on hardware
//
#include <hip/hip_runtime.h>
#include <hip/hip_bf16.h>

typedef __attribute__((ext_vector_type(16))) _Float16 v16h;
typedef __attribute__((ext_vector_type(8)))  float    v8f;

#define IN_F 8192
#define OUT_F 8192
#define TOKENS 4096
#define SCALE_BLOCKS (IN_F / 64)
#define O_CHUNKS 32
#define O_PER (OUT_F / O_CHUNKS)   // 256 rows of w1 per wave

__device__ __constant__ float NF4_TBL_C[16] = {
    -1.0f, -0.6961928009986877f, -0.5250730514526367f, -0.39491748809814453f,
    -0.28444138169288635f, -0.18477343022823334f, -0.09105003625154495f, 0.0f,
    0.07958029955625534f, 0.16093020141124725f, 0.24611230194568634f,
    0.33791524171829224f, 0.44070982933044434f, 0.5626170039176941f,
    0.7229568362236023f, 1.0f};

// Pass 0: zero the folded-weight vector v[IN_F] in workspace.
__global__ void nf4_zero(float* __restrict__ v) {
  v[blockIdx.x * 256 + threadIdx.x] = 0.0f;
}

// Pass A: v = w1_deq.T @ w2[0], via WMMA with w2 in row 0 of the A matrix.
// One wave (32 threads) per block; grid = (IN_F/16 i-tiles, O_CHUNKS o-chunks).
__global__ __launch_bounds__(32)
void nf4_fold_w(const int*   __restrict__ codes,   // [OUT_F, IN_F] values 0..15
                const float* __restrict__ scales,  // [OUT_F, IN_F/64]
                const float* __restrict__ w2,      // [1, OUT_F]
                float*       __restrict__ v) {     // [IN_F] accumulator
  __shared__ float tbl[16];
  const int lane = threadIdx.x;          // wave32: lane == threadIdx.x
  if (lane < 16) tbl[lane] = NF4_TBL_C[lane];
  __syncthreads();

  const int i0    = blockIdx.x * 16;     // 16-column tile of in_f
  const int oBase = blockIdx.y * O_PER;  // this wave's o-range
  const int half  = lane >> 4;           // 0: lanes 0-15, 1: lanes 16-31
  const int n     = lane & 15;           // matrix row (A) / column (B) id
  const int sblk  = i0 >> 6;             // 16-col tile lies inside one 64-block

  v8f c = {};                            // fp32 accumulator; row 0 holds v-partials
  for (int o = oBase; o < oBase + O_PER; o += 32) {
    // A (16x32 f16): row 0 = w2[o..o+31], rows 1..15 = 0.
    // Lane holding row 0 in half h supplies K = h*8..h*8+7 and 16+h*8..16+h*8+7.
    v16h a = {};
    if (n == 0) {
#pragma unroll
      for (int j = 0; j < 8; ++j) {
        a[j]     = (_Float16)w2[o + half * 8 + j];
        a[j + 8] = (_Float16)w2[o + 16 + half * 8 + j];
      }
    }
    // B (32x16 f16): lane = column i0+n, elements j -> K = half*8 + j (j<8), 16 + half*8 + (j-8) (j>=8).
    // Dequantize NF4: table lookup (LDS) * per-64-block scale, to f16.
    v16h b;
#pragma unroll
    for (int j = 0; j < 16; ++j) {
      const int row  = o + (j < 8 ? half * 8 + j : 16 + half * 8 + (j - 8));
      const int code = codes[row * IN_F + (i0 + n)];
      const float s  = scales[row * SCALE_BLOCKS + sblk];
      b[j] = (_Float16)(tbl[code & 15] * s);
    }
    // D = A*B + C : D[0,n] = sum_k w2[o+k] * w1deq[o+k, i0+n]
    c = __builtin_amdgcn_wmma_f32_16x16x32_f16(
        /*neg_a=*/false, a, /*neg_b=*/false, b,
        /*c_mod=*/(short)0, c, /*reuse_a=*/false, /*reuse_b=*/false);
  }
  // Row 0 of D sits in C VGPR0 on lanes 0..15 (N = lane).
  if (lane < 16) atomicAdd(&v[i0 + lane], c[0]);
}

// Pass B: out[t] = x[t,:] . v   (memory-bound GEMV, v is L2-resident)
__global__ __launch_bounds__(256)
void nf4_gemv(const float* __restrict__ x, const float* __restrict__ v,
              float* __restrict__ out) {
  __shared__ float red[256];
  const int row = blockIdx.x;
  const float4* x4 = (const float4*)(x + (size_t)row * IN_F);
  const float4* v4 = (const float4*)v;
  float s = 0.0f;
  for (int k = threadIdx.x; k < IN_F / 4; k += 256) {
    const float4 xa = x4[k];
    const float4 vb = v4[k];
    s += xa.x * vb.x + xa.y * vb.y + xa.z * vb.z + xa.w * vb.w;
  }
  red[threadIdx.x] = s;
  __syncthreads();
  for (int w = 128; w > 0; w >>= 1) {
    if (threadIdx.x < w) red[threadIdx.x] += red[threadIdx.x + w];
    __syncthreads();
  }
  if (threadIdx.x == 0) out[row] = red[0];
}

extern "C" void kernel_launch(void* const* d_in, const int* in_sizes, int n_in,
                              void* d_out, int out_size, void* d_ws, size_t ws_size,
                              hipStream_t stream) {
  const float* x      = (const float*)d_in[0];  // [4096, 8192] f32
  const int*   codes  = (const int*)  d_in[1];  // [8192, 8192] i32 (0..15)
  const float* scales = (const float*)d_in[2];  // [8192, 128] f32
  const float* w2     = (const float*)d_in[3];  // [1, 8192] f32
  float*       out    = (float*)d_out;          // [4096, 1] f32
  float*       v      = (float*)d_ws;           // [8192] f32 scratch

  nf4_zero<<<IN_F / 256, 256, 0, stream>>>(v);
  nf4_fold_w<<<dim3(IN_F / 16, O_CHUNKS), 32, 0, stream>>>(codes, scales, w2, v);
  nf4_gemv<<<TOKENS, 256, 0, stream>>>(x, v, out);
}
